// MultiHeadAttention_82016695484636
// MI455X (gfx1250) — matrix operand order repaired and checked by reading, not compiled, not run on hardware
//
#include <hip/hip_runtime.h>
#include <hip/hip_bf16.h>

// Graphormer-style MHA, fully fused on gfx1250 (MI455X, wave32, WMMA).
// Pipeline: cast/transpose -> edge table precontract -> QKV bf16 WMMA GEMMs
// -> fused flash-attention (WMMA QK^T + inline bias gathers + online softmax
// + WMMA P*V) -> output projection WMMA GEMM (f32 out).
// Workspace usage: ~22 MB of d_ws.

#define N_NODES 1024
#define IN_DIM  1024
#define H_HEADS 16
#define HD      64
#define L_PATH  4
#define D_EDGE  64
#define E_EDGES 16384

typedef __attribute__((ext_vector_type(16))) __bf16 v16bf_t;
typedef __attribute__((ext_vector_type(8)))  float  v8f_t;

union FragBF { uint4 u[2]; v16bf_t v; };

static __device__ __forceinline__ v8f_t wmma_bf16(const FragBF& a, const FragBF& b, v8f_t c) {
  // D(16x16,f32) = A(16x32,bf16) * B(32x16,bf16) + C
  return __builtin_amdgcn_wmma_f32_16x16x32_bf16(false, a.v, false, b.v, (short)0, c, false, false);
}

// ---------------------------------------------------------------------------
// Kernel 1: cast node features to bf16; cast + transpose the 4 weight matrices
// (Wt[out][in]) so WMMA B-fragments read two contiguous 16-byte k runs per lane.
// ---------------------------------------------------------------------------
__global__ __launch_bounds__(256) void cast_kernel(
    const float* __restrict__ X,
    const float* __restrict__ Wq, const float* __restrict__ Wk,
    const float* __restrict__ Wv, const float* __restrict__ Wo,
    __hip_bfloat16* __restrict__ Xb,
    __hip_bfloat16* __restrict__ WqT, __hip_bfloat16* __restrict__ WkT,
    __hip_bfloat16* __restrict__ WvT, __hip_bfloat16* __restrict__ WoT) {
  const int idx = blockIdx.x * 256 + threadIdx.x;           // < 1024*1024
  const int k = idx >> 10;                                   // input dim
  const int o = idx & 1023;                                  // output dim
  Xb[idx] = __float2bfloat16(X[idx]);
  const size_t t = (size_t)o * IN_DIM + k;
  WqT[t] = __float2bfloat16(Wq[idx]);
  WkT[t] = __float2bfloat16(Wk[idx]);
  WvT[t] = __float2bfloat16(Wv[idx]);
  WoT[t] = __float2bfloat16(Wo[idx]);
}

// ---------------------------------------------------------------------------
// Kernel 2: C[e, l, h] = pad_edge_feat[e] . edge_weight[l, h]   (E+1, L*H)
// edge_weight (16 KB) staged in LDS; row e == E is the zero pad row.
// ---------------------------------------------------------------------------
__global__ __launch_bounds__(256) void edge_table_kernel(
    const float* __restrict__ edge_feat, const float* __restrict__ edge_w,
    float* __restrict__ Ctab) {
  __shared__ float w[L_PATH * H_HEADS * D_EDGE];            // 16 KB
  const int t = threadIdx.x;
  for (int i = t; i < L_PATH * H_HEADS * D_EDGE; i += 256) w[i] = edge_w[i];
  __syncthreads();
  const int e  = blockIdx.x * 4 + (t >> 6);
  const int lh = t & 63;
  if (e >= E_EDGES + 1) return;
  float acc = 0.f;
  if (e < E_EDGES) {
    const float* ef = edge_feat + (size_t)e * D_EDGE;
    const float* wr = w + lh * D_EDGE;
#pragma unroll 8
    for (int d = 0; d < D_EDGE; ++d) acc += ef[d] * wr[d];
  }
  Ctab[(size_t)e * 64 + lh] = acc;                          // layout [E+1][L][H]
}

// ---------------------------------------------------------------------------
// Kernel 3: bf16 GEMM, C[1024x1024] = A[1024x1024] * Bt^T + bias.
// Block = 128 threads (4 waves), tile 64(M) x 64(N); wave owns a 16x64 strip.
// MODE 0: store bf16 row-major.  MODE 1: store bf16 transposed (for V^T).
// MODE 2: store f32 row-major (final output).
// ---------------------------------------------------------------------------
template <int MODE>
__global__ __launch_bounds__(128) void gemm64_kernel(
    const __hip_bfloat16* __restrict__ A, const __hip_bfloat16* __restrict__ Bt,
    const float* __restrict__ bias, void* __restrict__ Cout) {
  const int lane = threadIdx.x & 31;
  const int wv   = threadIdx.x >> 5;
  const int ln   = lane & 15;
  const int hf   = lane >> 4;
  const int row0 = blockIdx.y * 64 + wv * 16;
  const int col0 = blockIdx.x * 64;

  v8f_t acc[4] = {};
  const uint4* Arow = (const uint4*)(A + (size_t)(row0 + ln) * IN_DIM);
  const uint4* Br[4];
#pragma unroll
  for (int c = 0; c < 4; ++c)
    Br[c] = (const uint4*)(Bt + (size_t)(col0 + c * 16 + ln) * IN_DIM);

  for (int k0 = 0; k0 < IN_DIM; k0 += 32) {
    FragBF a;                                               // A: 16x32 bf16
    const int ai = (k0 >> 3) + hf;                          // K = 8*hf + j(mod 8) (+16)
    a.u[0] = Arow[ai];
    a.u[1] = Arow[ai + 2];
#pragma unroll
    for (int c = 0; c < 4; ++c) {
      FragBF b;                                             // B: 32x16 bf16 (from Wt)
      const int bi = (k0 >> 3) + hf;                        // K = 8*hf + j(mod 8) (+16)
      b.u[0] = Br[c][bi];
      b.u[1] = Br[c][bi + 2];
      acc[c] = wmma_bf16(a, b, acc[c]);
    }
  }
#pragma unroll
  for (int c = 0; c < 4; ++c) {
#pragma unroll
    for (int v = 0; v < 8; ++v) {                           // D: row = v+8*hf, col = ln
      const int row = row0 + v + 8 * hf;
      const int col = col0 + c * 16 + ln;
      const float val = acc[c][v] + bias[col];
      if (MODE == 0)
        ((__hip_bfloat16*)Cout)[(size_t)row * IN_DIM + col] = __float2bfloat16(val);
      else if (MODE == 1)
        ((__hip_bfloat16*)Cout)[(size_t)col * N_NODES + row] = __float2bfloat16(val);
      else
        ((float*)Cout)[(size_t)row * IN_DIM + col] = val;
    }
  }
}

// ---------------------------------------------------------------------------
// Kernel 4: fused flash attention with spatial + edge-path biases.
// Block = (head, 64-query slab), 4 waves; wave owns 16 queries of one head.
// Per 32-key tile: 4 WMMAs for S = Q K^T, inline bias gathers + mask, online
// softmax (shfl_xor width-16 row reductions), P transposed via per-wave LDS,
// 4 WMMAs for O += P V.
// ---------------------------------------------------------------------------
__global__ __launch_bounds__(128) void attn_kernel(
    const __hip_bfloat16* __restrict__ Qb,   // [N][IN_DIM] bf16
    const __hip_bfloat16* __restrict__ Kb,   // [N][IN_DIM] bf16
    const __hip_bfloat16* __restrict__ Vt,   // [IN_DIM][N] bf16 (transposed)
    const int*   __restrict__ sdist,         // [N][N]
    const int*   __restrict__ spaths,        // [N][N][L]
    const float* __restrict__ spatial,       // [L+1][H]
    const float* __restrict__ Ctab,          // [E+1][L][H]
    __hip_bfloat16* __restrict__ Ob) {       // [N][IN_DIM] bf16 attention out
  constexpr float SCALE = 0.125f;            // 1/sqrt(64)
  const int lane = threadIdx.x & 31;
  const int wv   = threadIdx.x >> 5;
  const int ln   = lane & 15;
  const int hf   = lane >> 4;
  const int h    = blockIdx.x;
  const int n0   = blockIdx.y * 64 + wv * 16;

  __shared__ __align__(16) __hip_bfloat16 Pbuf[4][16 * 32]; // 1 KB / wave
  __hip_bfloat16* Pw = &Pbuf[wv][0];

  // Q A-fragments for head-dim chunks d0=0 and d0=32 (loaded once)
  FragBF aq0, aq1;
  const uint4* Qrow = (const uint4*)(Qb + (size_t)(n0 + ln) * IN_DIM + h * HD);
  aq0.u[0] = Qrow[hf];     aq0.u[1] = Qrow[hf + 2];
  aq1.u[0] = Qrow[4 + hf]; aq1.u[1] = Qrow[6 + hf];

  float mst[8], lst[8];
#pragma unroll
  for (int v = 0; v < 8; ++v) { mst[v] = -1e30f; lst[v] = 0.f; }
  v8f_t o[4] = {};

  for (int m0 = 0; m0 < N_NODES; m0 += 32) {
    if (m0 + 32 < N_NODES) {                 // gfx1250 global_prefetch_b8
      __builtin_prefetch(Kb + (size_t)(m0 + 32 + ln) * IN_DIM + h * HD, 0, 1);
      __builtin_prefetch(Vt + (size_t)(h * HD + ln) * N_NODES + m0 + 32, 0, 1);
    }
    // ---- S = Q K^T for this 16x32 tile (two 16x16 column chunks) ----
    v8f_t s[2] = {};
#pragma unroll
    for (int c = 0; c < 2; ++c) {
      const uint4* Kr = (const uint4*)(Kb + (size_t)(m0 + c * 16 + ln) * IN_DIM + h * HD);
      FragBF b0, b1;
      b0.u[0] = Kr[hf];     b0.u[1] = Kr[hf + 2];       // d in [0,32)
      b1.u[0] = Kr[4 + hf]; b1.u[1] = Kr[6 + hf];       // d in [32,64)
      s[c] = wmma_bf16(aq0, b0, s[c]);
      s[c] = wmma_bf16(aq1, b1, s[c]);
    }
    // ---- per-element: scale + spatial bias + edge-path bias + mask ----
    float sv[2][8];
#pragma unroll
    for (int c = 0; c < 2; ++c) {
      const int m = m0 + c * 16 + ln;
#pragma unroll
      for (int v = 0; v < 8; ++v) {
        const int n = n0 + v + 8 * hf;
        const size_t pidx = (size_t)n * N_NODES + m;
        const int d = sdist[pidx];
        float val = -1e30f;                                     // masked (d <= 0)
        if (d > 0) {
          const int dc = d < L_PATH ? d : L_PATH;
          const int* pp = spaths + pidx * L_PATH;
          float ee = Ctab[(size_t)pp[0] * 64      + h]
                   + Ctab[(size_t)pp[1] * 64 + 16 + h]
                   + Ctab[(size_t)pp[2] * 64 + 32 + h]
                   + Ctab[(size_t)pp[3] * 64 + 48 + h];
          val = s[c][v] * SCALE + spatial[dc * H_HEADS + h] + ee;
        }
        sv[c][v] = val;
      }
    }
    // ---- online softmax update (rows live in 16-lane halves) ----
#pragma unroll
    for (int v = 0; v < 8; ++v) {
      float t = fmaxf(sv[0][v], sv[1][v]);
#pragma unroll
      for (int off = 1; off < 16; off <<= 1) t = fmaxf(t, __shfl_xor(t, off, 16));
      const float mnew = fmaxf(mst[v], t);
      const float sc = __expf(mst[v] - mnew);
      mst[v] = mnew;
      float rs = 0.f;
#pragma unroll
      for (int c = 0; c < 2; ++c) {
        const float p = __expf(sv[c][v] - mnew);
        sv[c][v] = p;
        rs += p;
      }
#pragma unroll
      for (int off = 1; off < 16; off <<= 1) rs += __shfl_xor(rs, off, 16);
      lst[v] = lst[v] * sc + rs;
#pragma unroll
      for (int c4 = 0; c4 < 4; ++c4) o[c4][v] *= sc;
    }
    // ---- transpose P (D-layout -> A-layout) through per-wave LDS ----
#pragma unroll
    for (int c = 0; c < 2; ++c)
#pragma unroll
      for (int v = 0; v < 8; ++v)
        Pw[(v + 8 * hf) * 32 + c * 16 + ln] = __float2bfloat16(sv[c][v]);
    // same-wave DS ops are in-order: reads below observe the stores above
    FragBF ap;
    const uint4* Pr = (const uint4*)(Pw + ln * 32 + 8 * hf);
    ap.u[0] = Pr[0];
    ap.u[1] = Pr[2];
    // ---- O += P * V  (V^T gives contiguous B-fragment loads) ----
#pragma unroll
    for (int c4 = 0; c4 < 4; ++c4) {
      const uint4* Vr = (const uint4*)(Vt + (size_t)(h * HD + c4 * 16 + ln) * N_NODES + m0);
      FragBF bv;
      bv.u[0] = Vr[hf];
      bv.u[1] = Vr[hf + 2];
      o[c4] = wmma_bf16(ap, bv, o[c4]);
    }
  }
  // ---- normalize and store attention output (bf16, row-major) ----
#pragma unroll
  for (int c4 = 0; c4 < 4; ++c4)
#pragma unroll
    for (int v = 0; v < 8; ++v) {
      const int n = n0 + v + 8 * hf;
      const int d = h * HD + c4 * 16 + ln;
      Ob[(size_t)n * IN_DIM + d] = __float2bfloat16(o[c4][v] / lst[v]);
    }
}

// ---------------------------------------------------------------------------
extern "C" void kernel_launch(void* const* d_in, const int* in_sizes, int n_in,
                              void* d_out, int out_size, void* d_ws, size_t ws_size,
                              hipStream_t stream) {
  (void)in_sizes; (void)n_in; (void)out_size; (void)ws_size;
  const float* node_feat = (const float*)d_in[0];
  const float* edge_feat = (const float*)d_in[1];
  // d_in[2] = distances (unused by reference), d_in[5] = edge_type (unused)
  const int*   sdist     = (const int*)d_in[3];
  const int*   spaths    = (const int*)d_in[4];
  const float* Wq = (const float*)d_in[6];  const float* bq = (const float*)d_in[7];
  const float* Wk = (const float*)d_in[8];  const float* bk = (const float*)d_in[9];
  const float* Wv = (const float*)d_in[10]; const float* bv = (const float*)d_in[11];
  const float* Wo = (const float*)d_in[12]; const float* bo = (const float*)d_in[13];
  const float* spatial = (const float*)d_in[14];
  const float* edge_w  = (const float*)d_in[15];

  char* ws = (char*)d_ws;
  const size_t MB = (size_t)1 << 20;
  __hip_bfloat16* Xb  = (__hip_bfloat16*)(ws + 0 * MB);   // 2 MB
  __hip_bfloat16* WqT = (__hip_bfloat16*)(ws + 2 * MB);   // 2 MB
  __hip_bfloat16* WkT = (__hip_bfloat16*)(ws + 4 * MB);
  __hip_bfloat16* WvT = (__hip_bfloat16*)(ws + 6 * MB);
  __hip_bfloat16* WoT = (__hip_bfloat16*)(ws + 8 * MB);
  __hip_bfloat16* Qb  = (__hip_bfloat16*)(ws + 10 * MB);
  __hip_bfloat16* Kb  = (__hip_bfloat16*)(ws + 12 * MB);
  __hip_bfloat16* Vt  = (__hip_bfloat16*)(ws + 14 * MB);
  __hip_bfloat16* Ob  = (__hip_bfloat16*)(ws + 16 * MB);
  float*          Ctab = (float*)(ws + 18 * MB);          // (E+1)*64*4 ~ 4.0 MB

  cast_kernel<<<(N_NODES * IN_DIM) / 256, 256, 0, stream>>>(
      node_feat, Wq, Wk, Wv, Wo, Xb, WqT, WkT, WvT, WoT);
  edge_table_kernel<<<(E_EDGES + 1 + 3) / 4, 256, 0, stream>>>(edge_feat, edge_w, Ctab);

  dim3 g(IN_DIM / 64, N_NODES / 64);
  gemm64_kernel<0><<<g, 128, 0, stream>>>(Xb, WqT, bq, Qb);
  gemm64_kernel<0><<<g, 128, 0, stream>>>(Xb, WkT, bk, Kb);
  gemm64_kernel<1><<<g, 128, 0, stream>>>(Xb, WvT, bv, Vt);   // store V transposed

  attn_kernel<<<dim3(H_HEADS, N_NODES / 64), 128, 0, stream>>>(
      Qb, Kb, Vt, sdist, spaths, spatial, Ctab, Ob);

  gemm64_kernel<2><<<g, 128, 0, stream>>>(Ob, WoT, bo, d_out); // f32 out + bias
}
